// DiffGLCM_86732569576023
// MI455X (gfx1250) — compile-verified, matrix operand order checked by reading, not run on hardware
//
#include <hip/hip_runtime.h>
#include <hip/hip_bf16.h>

typedef __attribute__((ext_vector_type(16))) _Float16 v16h;
typedef __attribute__((ext_vector_type(8)))  _Float16 v8h;
typedef __attribute__((ext_vector_type(8)))  float    v8f;

#define NGLV 64          // number of gray levels
#define IMG  256
#define TPB  32          // pixel-tile blocks per batch image

// Compute the 64 soft-level values T[c] for one pixel value x and store them
// into column `lane` of a channel-major LDS tile dst[64][32] (f16).
// T[0] = 1 - sig(10*(xn-2)); T[c] = sig(10*(xn-(c+1))) - sig(10*(xn-(c+2)));
// T[63] = sig(10*(xn-64)).  xn = x*64 + 1.
__device__ __forceinline__ void emit_levels(float x, float mask,
                                            _Float16 (*dst)[32], int lane) {
    float xn  = __builtin_fmaf(x, 64.0f, 1.0f);
    float a10 = 10.0f * xn;
    float prev = 1.0f;                     // boundary[0] = ones channel
    #pragma unroll 3
    for (int c = 0; c < 63; ++c) {
        // boundary[c+1] = sigmoid(a10 - 10*(c+2))
        float arg = (float)(10 * (c + 2)) - a10;     // -(t), sigmoid(t)=1/(1+e^-t)
        float sc  = __builtin_amdgcn_rcpf(1.0f + __expf(arg));
        dst[c][lane] = (_Float16)((prev - sc) * mask);
        prev = sc;
    }
    dst[63][lane] = (_Float16)(prev * mask);
}

// Load a 16x32 f16 A/B fragment (channel block `blk`) from a channel-major
// LDS tile, per the CDNA5 16-bit matrix VGPR layout:
//   lanes 0-15 : row = blk*16 + lane,    K = 0-7  (V0-3) and 16-23 (V4-7)
//   lanes 16-31: row = blk*16 + lane-16, K = 8-15 (V0-3) and 24-31 (V4-7)
__device__ __forceinline__ v16h load_frag(const _Float16 (*S)[32], int blk, int lane) {
    int row = blk * 16 + (lane & 15);
    int kb  = (lane >> 4) << 3;            // 0 or 8
    union { v16h v; v8h h[2]; } u;
    u.h[0] = *reinterpret_cast<const v8h*>(&S[row][kb]);
    u.h[1] = *reinterpret_cast<const v8h*>(&S[row][kb + 16]);
    return u.v;
}

__global__ void glcm_zero_kernel(float* __restrict__ out, int n) {
    int i = blockIdx.x * blockDim.x + threadIdx.x;
    if (i < n) out[i] = 0.0f;
}

__global__ __launch_bounds__(32)
void glcm_main_kernel(const float* __restrict__ x,
                      const int* __restrict__ p_offr,
                      const int* __restrict__ p_offc,
                      float* __restrict__ G) {
    __shared__ _Float16 As[NGLV][32];      // center levels, channel-major
    __shared__ _Float16 Bs[NGLV][32];      // periph levels, channel-major

    const int lane = threadIdx.x;          // 0..31 (wave32)
    const int b    = blockIdx.y;
    const int offr = *p_offr;
    const int offc = *p_offc;
    const int H    = IMG - offr;
    const int W    = IMG - offc;
    const int npix = H * W;
    const float* xb = x + (size_t)b * IMG * IMG;

    v8f acc[4][4] = {};                    // 64x64 f32 accumulators (4x4 WMMA tiles)

    for (int chunk = blockIdx.x; chunk * 32 < npix; chunk += gridDim.x) {
        int   p     = chunk * 32 + lane;
        float mask  = (p < npix) ? 1.0f : 0.0f;
        int   pc    = (p < npix) ? p : 0;
        int   h     = pc / W;
        int   w     = pc - h * W;
        float xc = xb[h * IMG + w];
        float xp = xb[(h + offr) * IMG + (w + offc)];

        emit_levels(xc, mask, As, lane);
        emit_levels(xp, mask, Bs, lane);

        // make this wave's LDS stores visible before fragment loads
        __asm__ volatile("s_wait_dscnt 0" ::: "memory");

        v16h af[4], bf[4];
        #pragma unroll
        for (int m = 0; m < 4; ++m) af[m] = load_frag(As, m, lane);
        #pragma unroll
        for (int n = 0; n < 4; ++n) bf[n] = load_frag(Bs, n, lane);

        #pragma unroll
        for (int m = 0; m < 4; ++m)
            #pragma unroll
            for (int n = 0; n < 4; ++n)
                acc[m][n] = __builtin_amdgcn_wmma_f32_16x16x32_f16(
                    /*neg_a=*/false, af[m], /*neg_b=*/false, bf[n],
                    /*c_mod=*/(short)0, acc[m][n],
                    /*reuse_a=*/false, /*reuse_b=*/false);

        // fragment loads done before next iteration overwrites LDS
        __asm__ volatile("s_wait_dscnt 0" ::: "memory");
    }

    // Scatter-accumulate this wave's 64x64 partial into global G[b].
    // C/D layout: VGPR r, lanes 0-15 -> M = r, lanes 16-31 -> M = r+8; N = lane&15.
    float* Gb = G + (size_t)b * NGLV * NGLV;
    const int rbase = (lane >> 4) * 8;
    const int col   = lane & 15;
    #pragma unroll
    for (int m = 0; m < 4; ++m)
        #pragma unroll
        for (int n = 0; n < 4; ++n)
            #pragma unroll
            for (int r = 0; r < 8; ++r) {
                int row = m * 16 + rbase + r;
                __hip_atomic_fetch_add(&Gb[row * NGLV + (n * 16 + col)],
                                       acc[m][n][r],
                                       __ATOMIC_RELAXED, __HIP_MEMORY_SCOPE_AGENT);
            }
}

__global__ __launch_bounds__(256)
void glcm_norm_kernel(float* __restrict__ out) {
    const int b   = blockIdx.x;
    const int tid = threadIdx.x;           // 256 threads, 4096 elems per batch
    float* Gb = out + (size_t)b * NGLV * NGLV;

    float v[16];
    float s = 0.0f;
    #pragma unroll
    for (int i = 0; i < 16; ++i) {
        v[i] = Gb[i * 256 + tid];
        s += v[i];
    }

    __shared__ float red[256];
    red[tid] = s;
    __syncthreads();
    #pragma unroll
    for (int off = 128; off > 0; off >>= 1) {
        if (tid < off) red[tid] += red[tid + off];
        __syncthreads();
    }
    float inv = 1.0f / red[0];

    #pragma unroll
    for (int i = 0; i < 16; ++i)
        Gb[i * 256 + tid] = v[i] * inv;
}

extern "C" void kernel_launch(void* const* d_in, const int* in_sizes, int n_in,
                              void* d_out, int out_size, void* d_ws, size_t ws_size,
                              hipStream_t stream) {
    const float* x      = (const float*)d_in[0];
    const int*   p_offr = (const int*)d_in[1];
    const int*   p_offc = (const int*)d_in[2];
    float*       out    = (float*)d_out;

    const int B = in_sizes[0] / (IMG * IMG);

    glcm_zero_kernel<<<(out_size + 255) / 256, 256, 0, stream>>>(out, out_size);
    glcm_main_kernel<<<dim3(TPB, B), 32, 0, stream>>>(x, p_offr, p_offc, out);
    glcm_norm_kernel<<<B, 256, 0, stream>>>(out);
}
